// Decoder_39487929319708
// MI455X (gfx1250) — matrix operand order repaired and checked by reading, not compiled, not run on hardware
//
#include <hip/hip_runtime.h>
#include <hip/hip_bf16.h>
#include <cstdint>

// ---------------------------------------------------------------------------
// Problem constants (match reference)
// ---------------------------------------------------------------------------
#define Bb   64
#define Tt   32
#define Ss   64
#define Vv   32000
#define Ee   512
#define Hh   1024

typedef __attribute__((ext_vector_type(16))) __bf16 v16bf;
typedef __attribute__((ext_vector_type(8)))  float  v8f;
typedef unsigned int u32x4 __attribute__((ext_vector_type(4)));
typedef int          i32x4 __attribute__((ext_vector_type(4)));
typedef int          i32x8 __attribute__((ext_vector_type(8)));

#if defined(__has_builtin)
#if __has_builtin(__builtin_amdgcn_tensor_load_to_lds) && \
    __has_builtin(__builtin_amdgcn_s_wait_tensorcnt)
#define USE_TDM 1
#endif
#endif
#ifndef USE_TDM
#define USE_TDM 0
#endif

#define BP_PITCH 65               // 64 cols + 1 pad DWORD (TDM pad feature)
#define BP_FLOATS (32 * BP_PITCH)

#if USE_TDM
// ---------------------------------------------------------------------------
// Issue one TDM 2D-tile load: 32 rows x 64 cols of fp32 from W (row pitch
// ldb elements) into LDS at lds_byte_addr, padded to a 65-DWORD LDS pitch.
// D# layout per CDNA5 ISA. Tracked by TENSORcnt.
// ---------------------------------------------------------------------------
__device__ __forceinline__ void tdm_load_panel(unsigned lds_byte_addr,
                                               const float* gptr, int ldb)
{
    unsigned long long ga = (unsigned long long)(uintptr_t)gptr;
    u32x4 g0;
    g0[0] = 1u;                                   // count=1 (valid user D#)
    g0[1] = lds_byte_addr;                        // lds_addr (bytes)
    g0[2] = (unsigned)(ga & 0xffffffffu);         // global_addr[31:0]
    g0[3] = (unsigned)((ga >> 32) & 0x1ffffffu)   // global_addr[56:32]
          | (2u << 30);                           // type = 2 ("image")
    i32x8 g1;
    g1[0] = (2 << 16)      // data_size = 4 bytes
          | (1 << 20)      // pad_enable
          | (5 << 22)      // pad_interval: 64 DWORDs
          | (0 << 25);     // pad_amount: 1 DWORD  -> LDS pitch 65 DWORDs
    g1[1] = (ldb & 0xffff) << 16;                       // tensor_dim0[15:0]
    g1[2] = (int)(((unsigned)ldb >> 16) | (0xffffu << 16)); // dim0 hi | dim1 lo
    g1[3] = (64 << 16);                                 // dim1 hi=0 | tile_dim0=64
    g1[4] = 32;                                         // tile_dim1=32, tile_dim2=0
    g1[5] = ldb;                                        // tensor_dim0_stride[31:0]
    g1[6] = 0;                                          // stride hi | dim1_stride lo
    g1[7] = 0;
    i32x4 gz = {0, 0, 0, 0};
#if defined(__clang_major__) && (__clang_major__ >= 23)
    i32x8 gz8 = {0, 0, 0, 0, 0, 0, 0, 0};
    __builtin_amdgcn_tensor_load_to_lds(g0, g1, gz, gz, gz8, 0);
#else
    __builtin_amdgcn_tensor_load_to_lds(g0, g1, gz, gz, 0);
#endif
}
#endif // USE_TDM

// ---------------------------------------------------------------------------
// Generic WMMA GEMM:  C[M,N] (+)= A[M,K] @ W[K,N]
//   one wave per 16x64 output tile: 4 independent 16x16 bf16 WMMA
//   accumulators (pipelined v_wmma, 4x A reuse). B panels (32x64 fp32) are
//   DMA'd by the Tensor Data Mover into double-buffered LDS and consumed via
//   bank-conflict-free ds_load; A rows are direct b128 global loads.
// mode: 0 = acc from zero, 1 = acc from bias[n] broadcast, 2 = acc from C
// rowIdx: optional gather for A rows (embedding lookup): row = rowIdx[m*rs+ro]
// Requires: M % 16 == 0, N % 64 == 0, K % 32 == 0.
// ---------------------------------------------------------------------------
__global__ __launch_bounds__(32)
void wmma_gemm_bf16(const float* __restrict__ A, int lda,
                    const float* __restrict__ W, int ldb,
                    float* __restrict__ C, int ldc,
                    int K, int mode, const float* __restrict__ bias,
                    const int* __restrict__ rowIdx, int rowStride, int rowOff)
{
    const int lane = threadIdx.x;          // 0..31
    const int r16  = lane & 15;
    const int half = lane >> 4;            // 0 or 1
    const int n0   = blockIdx.x * 64;      // N supertile base
    const int n    = n0 + r16;             // column of subtile 0
    const int mrow = blockIdx.y * 16 + r16;    // A row owned by this lane

    size_t arowi = rowIdx ? (size_t)rowIdx[(size_t)mrow * rowStride + rowOff]
                          : (size_t)mrow;
    const float* __restrict__ arow = A + arowi * (size_t)lda;

    v8f acc[4];
    if (mode == 1) {
        #pragma unroll
        for (int j = 0; j < 4; ++j) {
            float bv = bias[n + 16 * j];
            #pragma unroll
            for (int i = 0; i < 8; ++i) acc[j][i] = bv;
        }
    } else if (mode == 2) {
        #pragma unroll
        for (int j = 0; j < 4; ++j)
            #pragma unroll
            for (int i = 0; i < 8; ++i)
                acc[j][i] = C[(size_t)(blockIdx.y * 16 + half * 8 + i) * ldc
                              + n + 16 * j];
    } else {
        #pragma unroll
        for (int j = 0; j < 4; ++j)
            #pragma unroll
            for (int i = 0; i < 8; ++i) acc[j][i] = 0.0f;
    }

#if USE_TDM
    __shared__ float bpanel[2][BP_FLOATS];
    const unsigned lds0 = (unsigned)(uintptr_t)&bpanel[0][0];
    const unsigned lds1 = (unsigned)(uintptr_t)&bpanel[1][0];
    // prologue: DMA first B panel (rows 0..31, cols n0..n0+63)
    tdm_load_panel(lds0, W + n0, ldb);
#endif

    for (int k0 = 0; k0 < K; k0 += 32) {
        // ---- A tile 16x32 bf16 (shared by all 4 subtiles): lane holds row
        //      (lane&15); elem e is K = half*16 + e (matches B order)
        const float* __restrict__ ap = arow + k0 + half * 16;
        v16bf a;
        #pragma unroll
        for (int e = 0; e < 8; ++e) a[e]     = (__bf16)ap[e];
        #pragma unroll
        for (int e = 0; e < 8; ++e) a[8 + e] = (__bf16)ap[8 + e];

        v16bf b[4];
#if USE_TDM
        const int  buf  = (k0 >> 5) & 1;
        const bool more = (k0 + 32) < K;
        if (more)   // DMA next panel into the other buffer while we compute
            tdm_load_panel(buf ? lds0 : lds1,
                           W + (size_t)(k0 + 32) * ldb + n0, ldb);
        // TENSORcnt is in-order per wave: <=1 outstanding => current panel done
        if (more) __builtin_amdgcn_s_wait_tensorcnt(1);
        else      __builtin_amdgcn_s_wait_tensorcnt(0);
        asm volatile("" ::: "memory");   // don't reorder LDS reads vs DMA wait

        // ---- B tiles 32x16 bf16 from LDS: lane holds one column; elem e is
        //      K = half*16 + e; pitch 65 => bank = (row+col)%64, conflict-free
        const float* __restrict__ bl = &bpanel[buf][0] + half * 16 * BP_PITCH + r16;
        #pragma unroll
        for (int j = 0; j < 4; ++j)
            #pragma unroll
            for (int e = 0; e < 16; ++e)
                b[j][e] = (__bf16)bl[e * BP_PITCH + 16 * j];
#else
        const float* __restrict__ bp = W + (size_t)(k0 + half * 16) * ldb + n;
        if (k0 + 32 < K)
            __builtin_prefetch(bp + (size_t)32 * ldb, 0, 1);
        #pragma unroll
        for (int j = 0; j < 4; ++j)
            #pragma unroll
            for (int e = 0; e < 16; ++e)
                b[j][e] = (__bf16)bp[(size_t)e * ldb + 16 * j];
#endif

        // 4 independent accumulators -> WMMAs pipeline without RAW stalls
        #pragma unroll
        for (int j = 0; j < 4; ++j)
            acc[j] = __builtin_amdgcn_wmma_f32_16x16x32_bf16(
                         false, a, false, b[j], (short)0, acc[j], false, false);
    }

    #pragma unroll
    for (int j = 0; j < 4; ++j)
        #pragma unroll
        for (int i = 0; i < 8; ++i)
            C[(size_t)(blockIdx.y * 16 + half * 8 + i) * ldc + n + 16 * j] =
                acc[j][i];
}

// ---------------------------------------------------------------------------
// LSTM gate activation: z[B,4H] (i,f,g,o) + c -> new h, c
// ---------------------------------------------------------------------------
__device__ __forceinline__ float sigf(float x) { return 1.0f / (1.0f + expf(-x)); }

__global__ __launch_bounds__(256)
void lstm_act(const float* __restrict__ z, float* __restrict__ h,
              float* __restrict__ c)
{
    int idx = blockIdx.x * blockDim.x + threadIdx.x;   // 0 .. B*H-1
    int b = idx >> 10;            // / Hh
    int u = idx & (Hh - 1);
    const float* zr = z + (size_t)b * 4 * Hh;
    float ig = sigf(zr[u]);
    float fg = sigf(zr[Hh + u]);
    float gg = tanhf(zr[2 * Hh + u]);
    float og = sigf(zr[3 * Hh + u]);
    float cn = fg * c[idx] + ig * gg;
    c[idx] = cn;
    h[idx] = og * tanhf(cn);
}

// ---------------------------------------------------------------------------
// Bahdanau score: score[b,s] = sum_u v[u]*tanh(keys[b,s,u] + q[b,u])
// one block of 256 per (b,s) pair
// ---------------------------------------------------------------------------
__global__ __launch_bounds__(256)
void attn_score(const float* __restrict__ keys, const float* __restrict__ q,
                const float* __restrict__ v, float* __restrict__ score)
{
    int bs = blockIdx.x;          // b*S + s
    int b  = bs / Ss;
    const float* kr = keys + (size_t)bs * Hh;
    const float* qr = q + (size_t)b * Hh;
    float p = 0.0f;
    for (int u = threadIdx.x; u < Hh; u += 256)
        p += v[u] * tanhf(kr[u] + qr[u]);
    __shared__ float red[256];
    red[threadIdx.x] = p;
    __syncthreads();
    for (int s = 128; s > 0; s >>= 1) {
        if (threadIdx.x < s) red[threadIdx.x] += red[threadIdx.x + s];
        __syncthreads();
    }
    if (threadIdx.x == 0) score[bs] = red[0];
}

// ---------------------------------------------------------------------------
// softmax over S=64 + context: ctx[b,:] = sum_s align[b,s] * memory[b,s,:]
// one block of 256 per batch row
// ---------------------------------------------------------------------------
__global__ __launch_bounds__(256)
void attn_ctx(const float* __restrict__ score, const float* __restrict__ memory,
              float* __restrict__ ctx)
{
    int b = blockIdx.x;
    __shared__ float al[Ss];
    if (threadIdx.x < Ss) al[threadIdx.x] = score[b * Ss + threadIdx.x];
    __syncthreads();
    if (threadIdx.x == 0) {
        float mx = al[0];
        for (int s = 1; s < Ss; ++s) mx = fmaxf(mx, al[s]);
        float sum = 0.0f;
        for (int s = 0; s < Ss; ++s) { al[s] = expf(al[s] - mx); sum += al[s]; }
        float inv = 1.0f / sum;
        for (int s = 0; s < Ss; ++s) al[s] *= inv;
    }
    __syncthreads();
    const float* mb = memory + (size_t)b * Ss * Hh;
    for (int m = threadIdx.x; m < Hh; m += 256) {
        float acc = 0.0f;
        for (int s = 0; s < Ss; ++s)
            acc += al[s] * mb[(size_t)s * Hh + m];
        ctx[(size_t)b * Hh + m] = acc;
    }
}

// ---------------------------------------------------------------------------
// kernel_launch
// ---------------------------------------------------------------------------
extern "C" void kernel_launch(void* const* d_in, const int* in_sizes, int n_in,
                              void* d_out, int out_size, void* d_ws, size_t ws_size,
                              hipStream_t stream)
{
    (void)in_sizes; (void)n_in; (void)out_size; (void)ws_size;

    const int*   tokens = (const int*)  d_in[0];
    const float* memory = (const float*)d_in[1];
    const float* enc_h  = (const float*)d_in[2];
    const float* enc_c  = (const float*)d_in[3];
    const float* emb    = (const float*)d_in[4];
    const float* k0     = (const float*)d_in[5];
    const float* r0     = (const float*)d_in[6];
    const float* b0     = (const float*)d_in[7];
    const float* k1     = (const float*)d_in[8];
    const float* r1     = (const float*)d_in[9];
    const float* b1     = (const float*)d_in[10];
    const float* Wq     = (const float*)d_in[11];
    const float* Wk     = (const float*)d_in[12];
    const float* vvec   = (const float*)d_in[13];
    const float* Wa     = (const float*)d_in[14];
    const float* Wfc    = (const float*)d_in[15];
    const float* bfc    = (const float*)d_in[16];
    float*       logits = (float*)d_out;

    // workspace carve-up (fp32)
    float* ws    = (float*)d_ws;
    float* keys  = ws;                                // B*S*H   = 4,194,304
    float* z0    = keys  + (size_t)Bb * Ss * Hh;      // B*4H    =   262,144
    float* z1    = z0    + (size_t)Bb * 4 * Hh;
    float* h0    = z1    + (size_t)Bb * 4 * Hh;       // B*H each
    float* c0    = h0    + (size_t)Bb * Hh;
    float* h1    = c0    + (size_t)Bb * Hh;
    float* c1    = h1    + (size_t)Bb * Hh;
    float* q     = c1    + (size_t)Bb * Hh;
    float* ctx   = q     + (size_t)Bb * Hh;
    float* score = ctx   + (size_t)Bb * Hh;           // B*S = 4096
    float* outs  = score + (size_t)Bb * Ss;           // B*T*H = 2,097,152

    const dim3 wv(32, 1, 1);
    const size_t BH = (size_t)Bb * Hh * sizeof(float);

    // keys = memory @ Wk  : [B*S, H] x [H, H]
    wmma_gemm_bf16<<<dim3(Hh / 64, (Bb * Ss) / 16), wv, 0, stream>>>(
        memory, Hh, Wk, Hh, keys, Hh, Hh, 0, nullptr, nullptr, 0, 0);

    // init recurrent state
    hipMemcpyAsync(h0, enc_h, BH, hipMemcpyDeviceToDevice, stream);
    hipMemcpyAsync(c0, enc_c, BH, hipMemcpyDeviceToDevice, stream);
    hipMemcpyAsync(h1, enc_h, BH, hipMemcpyDeviceToDevice, stream);
    hipMemcpyAsync(c1, enc_c, BH, hipMemcpyDeviceToDevice, stream);

    const dim3 gZ(4 * Hh / 64, Bb / 16);   // [64, 4096] tiles
    const dim3 gH(Hh / 64, Bb / 16);       // [64, 1024] tiles

    for (int t = 0; t < Tt; ++t) {
        // z0 = b0 + emb[tokens[:,t]] @ k0[0:E]  (gathered A rows)
        wmma_gemm_bf16<<<gZ, wv, 0, stream>>>(
            emb, Ee, k0, 4 * Hh, z0, 4 * Hh, Ee, 1, b0, tokens, Tt, t);
        // z0 += attn_{t-1} @ k0[E:E+H]   (attn_0 == 0 -> skip at t==0)
        if (t > 0)
            wmma_gemm_bf16<<<gZ, wv, 0, stream>>>(
                outs + (size_t)(t - 1) * Hh, Tt * Hh,
                k0 + (size_t)Ee * 4 * Hh, 4 * Hh, z0, 4 * Hh, Hh, 2,
                nullptr, nullptr, 0, 0);
        // z0 += h0 @ r0
        wmma_gemm_bf16<<<gZ, wv, 0, stream>>>(
            h0, Hh, r0, 4 * Hh, z0, 4 * Hh, Hh, 2, nullptr, nullptr, 0, 0);
        lstm_act<<<(Bb * Hh) / 256, 256, 0, stream>>>(z0, h0, c0);

        // z1 = b1 + h0 @ k1 + h1 @ r1
        wmma_gemm_bf16<<<gZ, wv, 0, stream>>>(
            h0, Hh, k1, 4 * Hh, z1, 4 * Hh, Hh, 1, b1, nullptr, 0, 0);
        wmma_gemm_bf16<<<gZ, wv, 0, stream>>>(
            h1, Hh, r1, 4 * Hh, z1, 4 * Hh, Hh, 2, nullptr, nullptr, 0, 0);
        lstm_act<<<(Bb * Hh) / 256, 256, 0, stream>>>(z1, h1, c1);

        // q = h1 @ Wq
        wmma_gemm_bf16<<<gH, wv, 0, stream>>>(
            h1, Hh, Wq, Hh, q, Hh, Hh, 0, nullptr, nullptr, 0, 0);

        // Bahdanau attention
        attn_score<<<Bb * Ss, 256, 0, stream>>>(keys, q, vvec, score);
        attn_ctx<<<Bb, 256, 0, stream>>>(score, memory, ctx);

        // attn_t = [h1, ctx] @ Wa   -> written straight into outs[:, t, :]
        wmma_gemm_bf16<<<gH, wv, 0, stream>>>(
            h1, Hh, Wa, Hh, outs + (size_t)t * Hh, Tt * Hh, Hh, 0,
            nullptr, nullptr, 0, 0);
        wmma_gemm_bf16<<<gH, wv, 0, stream>>>(
            ctx, Hh, Wa + (size_t)Hh * Hh, Hh, outs + (size_t)t * Hh, Tt * Hh,
            Hh, 2, nullptr, nullptr, 0, 0);
    }

    // logits = outs @ Wfc + bfc : [B*T, H] x [H, V]
    wmma_gemm_bf16<<<dim3(Vv / 64, (Bb * Tt) / 16), wv, 0, stream>>>(
        outs, Hh, Wfc, Vv, logits, Vv, Hh, 1, bfc, nullptr, 0, 0);
}
